// SelfAttentionBase_8761733284047
// MI455X (gfx1250) — matrix operand order repaired and checked by reading, not compiled, not run on hardware
//
#include <hip/hip_runtime.h>

typedef unsigned short u16;
typedef __attribute__((ext_vector_type(16))) __bf16 bf16x16;
typedef __attribute__((ext_vector_type(8)))  float  f32x8;
typedef __attribute__((ext_vector_type(8)))  unsigned short u16x8;
typedef __attribute__((ext_vector_type(4)))  unsigned int u32x4;
typedef __attribute__((ext_vector_type(8)))  unsigned int u32x8;

#define B_TOTAL 32768
#define S_LEN 3
#define D_IN 512
#define F_DIM 512
#define H_HEADS 8
#define DH 64
#define NB 16                 // batches per workgroup
#define M_ROWS (NB * S_LEN)   // 48 rows per WG
#define TILE_ELEMS (M_ROWS * D_IN)   // 24576
#define LDS_STRIDE 520        // 512 + 8 u16 pad -> conflict-free column access
#define WMAT_ELEMS (512 * 512)

__device__ __forceinline__ u16 f2bf(float f) {
    unsigned u = __float_as_uint(f);
    u = u + 0x7FFFu + ((u >> 16) & 1u);   // round-to-nearest-even
    return (u16)(u >> 16);
}
__device__ __forceinline__ float bf2f(u16 h) {
    return __uint_as_float(((unsigned)h) << 16);
}
__device__ __forceinline__ unsigned lds_off(const void* p) {
    // generic pointers to LDS carry the allocation-relative offset in the low 32 bits
    return (unsigned)(unsigned long long)p;
}

union BF16Frag { u16x8 h[2]; bf16x16 v; };

__device__ __forceinline__ bf16x16 load_frag(const u16* p0, const u16* p1) {
    BF16Frag f;
    f.h[0] = *(const u16x8*)p0;
    f.h[1] = *(const u16x8*)p1;
    return f.v;
}

// ---- TDM: DMA one contiguous [48 x 512] fp32 tile global -> LDS ----------
__device__ __forceinline__ void tdm_load_tile(const float* __restrict__ gsrc,
                                              unsigned lds_byte_off) {
    unsigned long long ga = (unsigned long long)gsrc;
    u32x4 g0;
    g0[0] = 0x1u;                                  // count=1, user descriptor
    g0[1] = lds_byte_off;                          // lds_addr (bytes)
    g0[2] = (unsigned)ga;                          // global_addr[31:0]
    g0[3] = (unsigned)((ga >> 32) & 0x01FFFFFFu)   // global_addr[56:32]
            | 0x80000000u;                         // type = 2 ("image")
    u32x8 g1;
    g1[0] = 0x2u << 16;                            // data_size = 4 bytes
    g1[1] = (unsigned)TILE_ELEMS << 16;            // tensor_dim0[15:0]
    g1[2] = 0x1u << 16;                            // tensor_dim0 hi=0, tensor_dim1 = 1
    g1[3] = (unsigned)TILE_ELEMS << 16;            // tile_dim0 = 24576
    g1[4] = 0x1u;                                  // tile_dim1 = 1, tile_dim2 = 0
    g1[5] = (unsigned)TILE_ELEMS;                  // tensor_dim0_stride lo32
    g1[6] = (unsigned)TILE_ELEMS << 16;            // stride hi=0; tensor_dim1_stride lo16
    g1[7] = 0;                                     // tensor_dim1_stride hi
    asm volatile("tensor_load_to_lds %0, %1" :: "s"(g0), "s"(g1) : "memory");
}

// ---- stage: LDS fp32 tile -> LDS bf16 tile -------------------------------
__device__ __forceinline__ void convert_tile(const float* sF, u16* dst, int tid) {
#pragma unroll 1
    for (int i = 0; i < 24; ++i) {
        int e4 = i * 256 + tid;
        float4 val = ((const float4*)sF)[e4];
        int e = e4 * 4;
        int row = e >> 9, c = e & 511;
        union { u16 s[4]; uint2 u; } pk;
        pk.s[0] = f2bf(val.x); pk.s[1] = f2bf(val.y);
        pk.s[2] = f2bf(val.z); pk.s[3] = f2bf(val.w);
        *(uint2*)(dst + row * LDS_STRIDE + c) = pk.u;   // 8B aligned
    }
}

// ---- WMMA GEMM: [48 x 512] (LDS bf16) @ [512 x 512] (global bf16, N-major)
template<bool TO_LDS>
__device__ __forceinline__ void gemm_tile(
    const u16* __restrict__ Bt,     // bf16 weights, layout [N][K]
    const float* __restrict__ bias,
    const u16* sA,                  // LDS A [48][LDS_STRIDE]
    u16* sOut,                      // LDS bf16 out (TO_LDS)
    float* __restrict__ gOut)       // global fp32 out (!TO_LDS), row stride 512
{
    const int lane = threadIdx.x & 31;
    const int wave = threadIdx.x >> 5;
    const int col  = lane & 15;
    const int hi   = lane >> 4;     // 0 or 1 (K-half select)

    for (int nt = wave; nt < 32; nt += 8) {
        f32x8 acc0 = {}; f32x8 acc1 = {}; f32x8 acc2 = {};
        const u16* bp = Bt + (nt * 16 + col) * 512 + hi * 8;
        const u16* ap = sA + col * LDS_STRIDE + hi * 8;
        if (nt + 8 < 32)
            __builtin_prefetch(Bt + ((nt + 8) * 16 + col) * 512, 0, 0);
#pragma unroll 4
        for (int k0 = 0; k0 < 512; k0 += 32) {
            bf16x16 bf = load_frag(bp + k0, bp + k0 + 16);
            bf16x16 a0 = load_frag(ap + k0,                     ap + k0 + 16);
            bf16x16 a1 = load_frag(ap + 16 * LDS_STRIDE + k0,   ap + 16 * LDS_STRIDE + k0 + 16);
            bf16x16 a2 = load_frag(ap + 32 * LDS_STRIDE + k0,   ap + 32 * LDS_STRIDE + k0 + 16);
            acc0 = __builtin_amdgcn_wmma_f32_16x16x32_bf16(false, a0, false, bf, (short)0, acc0, false, false);
            acc1 = __builtin_amdgcn_wmma_f32_16x16x32_bf16(false, a1, false, bf, (short)0, acc1, false, false);
            acc2 = __builtin_amdgcn_wmma_f32_16x16x32_bf16(false, a2, false, bf, (short)0, acc2, false, false);
        }
        const float bb = bias[nt * 16 + col];
        const int c = nt * 16 + col;
        auto store_acc = [&](const f32x8& a, int mbase) {
#pragma unroll
            for (int i = 0; i < 8; ++i) {
                float vv = a[i] + bb;
                int row = mbase + hi * 8 + i;
                if constexpr (TO_LDS) sOut[row * LDS_STRIDE + c] = f2bf(vv);
                else                  gOut[row * 512 + c] = vv;
            }
        };
        store_acc(acc0, 0); store_acc(acc1, 16); store_acc(acc2, 32);
    }
}

// ---- scores + softmax: one thread per (batch,head) -----------------------
__device__ __forceinline__ void softmax_scores(
    const int* sMask, const u16* sQ, const u16* sK, float* sP, int tid)
{
    const int lb = tid >> 3, h = tid & 7;
    float pen[3];
#pragma unroll
    for (int s = 0; s < 3; ++s) pen[s] = (float)sMask[lb * 3 + s] * 1e9f;

#pragma unroll
    for (int s1 = 0; s1 < 3; ++s1) {
        float sc[3];
        const u16* qr = sQ + (lb * 3 + s1) * LDS_STRIDE + h * DH;
#pragma unroll
        for (int s2 = 0; s2 < 3; ++s2) {
            const u16* kr = sK + (lb * 3 + s2) * LDS_STRIDE + h * DH;
            float d = 0.f;
#pragma unroll 16
            for (int cidx = 0; cidx < DH; ++cidx)
                d += bf2f(qr[cidx]) * bf2f(kr[cidx]);
            sc[s2] = d * 0.125f - pen[s1] - pen[s2];
        }
        float mx = fmaxf(sc[0], fmaxf(sc[1], sc[2]));
        float e0 = __expf(sc[0] - mx), e1 = __expf(sc[1] - mx), e2 = __expf(sc[2] - mx);
        float inv = 1.f / (e0 + e1 + e2);
        float* p = sP + ((lb * 8 + h) * 3 + s1) * 3;
        p[0] = e0 * inv; p[1] = e1 * inv; p[2] = e2 * inv;
    }
}

// ---- attn = P @ Vh (tiny: 3x3 per head) ----------------------------------
__device__ __forceinline__ void attn_combine(
    const float* sP, const u16* sV, u16* sOut, int tid)
{
#pragma unroll 1
    for (int i = 0; i < 96; ++i) {
        int e = i * 256 + tid;
        int r = e >> 9;           // 0..47
        int f = e & 511;
        int lb = r / 3, s1 = r - lb * 3;
        int h = f >> 6;
        const float* p = sP + ((lb * 8 + h) * 3 + s1) * 3;
        float acc = 0.f;
#pragma unroll
        for (int s2 = 0; s2 < 3; ++s2)
            acc += p[s2] * bf2f(sV[(lb * 3 + s2) * LDS_STRIDE + f]);
        sOut[r * LDS_STRIDE + f] = f2bf(acc);
    }
}

// ---- prep: fp32 [K][N] weights -> bf16 [N][K] in workspace ---------------
__global__ __launch_bounds__(256) void prep_weights(
    const float* __restrict__ Wq, const float* __restrict__ Wk,
    const float* __restrict__ Wv, const float* __restrict__ Wo,
    u16* __restrict__ outw)
{
    int idx = blockIdx.x * 256 + threadIdx.x;   // 0 .. 262143
    int n = idx >> 9, kk = idx & 511;
    int src = kk * 512 + n;
    outw[0 * WMAT_ELEMS + idx] = f2bf(Wq[src]);
    outw[1 * WMAT_ELEMS + idx] = f2bf(Wk[src]);
    outw[2 * WMAT_ELEMS + idx] = f2bf(Wv[src]);
    outw[3 * WMAT_ELEMS + idx] = f2bf(Wo[src]);
}

// ---- fused attention: one WG per 16 batches ------------------------------
__global__ __launch_bounds__(256) void attn_fused(
    const float* __restrict__ q, const float* __restrict__ k, const float* __restrict__ v,
    const int* __restrict__ mask,
    const float* __restrict__ bq, const float* __restrict__ bk,
    const float* __restrict__ bvv, const float* __restrict__ bo,
    const u16* __restrict__ Wt,     // [4][512][512] bf16 N-major
    float* __restrict__ out)
{
    __shared__ float sF[TILE_ELEMS];           // TDM fp32 staging (96 KB)
    __shared__ u16  sA[M_ROWS * LDS_STRIDE];   // bf16 input (q -> k -> v)
    __shared__ u16  sB[M_ROWS * LDS_STRIDE];   // Qh, later Vh
    __shared__ u16  sC[M_ROWS * LDS_STRIDE];   // Kh, later attn
    __shared__ float sP[NB * H_HEADS * S_LEN * S_LEN];
    __shared__ __align__(16) int sMask[M_ROWS];

    const int tid  = threadIdx.x;
    const int wv   = tid >> 5;
    const int lane = tid & 31;
    const long base = (long)blockIdx.x * TILE_ELEMS;
    const int batch0 = blockIdx.x * NB;

    // mask rows -> LDS via async copy (ASYNCcnt path), 12 lanes x b128
    if (wv == 1) {
        if (lane < 12) {
            unsigned long long ga = (unsigned long long)(mask + batch0 * 3 + lane * 4);
            unsigned lo = lds_off(sMask) + lane * 16;
            asm volatile("global_load_async_to_lds_b128 %0, %1, off"
                         :: "v"(lo), "v"(ga) : "memory");
        }
        asm volatile("s_wait_asynccnt 0" ::: "memory");
    }

    // q tile via Tensor Data Mover
    if (wv == 0) {
        tdm_load_tile(q + base, lds_off(sF));
        __builtin_amdgcn_s_wait_tensorcnt(0);
    }
    __syncthreads();
    convert_tile(sF, sA, tid);                                      // sA = q bf16
    __syncthreads();

    if (wv == 0) tdm_load_tile(k + base, lds_off(sF));              // DMA overlaps GEMM
    gemm_tile<true>(Wt + 0 * WMAT_ELEMS, bq, sA, sB, nullptr);      // Qh -> sB
    if (wv == 0) __builtin_amdgcn_s_wait_tensorcnt(0);
    __syncthreads();
    convert_tile(sF, sA, tid);                                      // sA = k bf16
    __syncthreads();

    if (wv == 0) tdm_load_tile(v + base, lds_off(sF));              // DMA overlaps GEMM
    gemm_tile<true>(Wt + 1 * WMAT_ELEMS, bk, sA, sC, nullptr);      // Kh -> sC
    if (wv == 0) __builtin_amdgcn_s_wait_tensorcnt(0);
    __syncthreads();

    convert_tile(sF, sA, tid);                                      // sA = v bf16
    if (tid < NB * H_HEADS)
        softmax_scores(sMask, sB, sC, sP, tid);
    __syncthreads();

    gemm_tile<true>(Wt + 2 * WMAT_ELEMS, bvv, sA, sB, nullptr);     // Vh -> sB (Qh dead)
    __syncthreads();

    attn_combine(sP, sB, sC, tid);                                  // attn -> sC (Kh dead)
    __syncthreads();

    gemm_tile<false>(Wt + 3 * WMAT_ELEMS, bo, sC, nullptr, out + base);
}

extern "C" void kernel_launch(void* const* d_in, const int* in_sizes, int n_in,
                              void* d_out, int out_size, void* d_ws, size_t ws_size,
                              hipStream_t stream) {
    const float* q   = (const float*)d_in[0];
    const float* k   = (const float*)d_in[1];
    const float* v   = (const float*)d_in[2];
    const int*   msk = (const int*)d_in[3];
    const float* Wq  = (const float*)d_in[4];
    const float* bq  = (const float*)d_in[5];
    const float* Wk  = (const float*)d_in[6];
    const float* bk  = (const float*)d_in[7];
    const float* Wv  = (const float*)d_in[8];
    const float* bv  = (const float*)d_in[9];
    const float* Wo  = (const float*)d_in[10];
    const float* bo  = (const float*)d_in[11];
    u16*   Wt  = (u16*)d_ws;           // 4 * 512*512 * 2B = 2 MB
    float* out = (float*)d_out;

    prep_weights<<<WMAT_ELEMS / 256, 256, 0, stream>>>(Wq, Wk, Wv, Wo, Wt);
    attn_fused<<<B_TOTAL / NB, 256, 0, stream>>>(q, k, v, msk, bq, bk, bv, bo, Wt, out);
}
